// MoELayer_38946763440631
// MI455X (gfx1250) — matrix operand order repaired and checked by reading, not compiled, not run on hardware
//
#include <hip/hip_runtime.h>

// ---------------- problem constants ----------------
constexpr int kE = 8;      // experts
constexpr int kD = 512;    // embed
constexpr int kI = 2048;   // inter
constexpr int kN = 4096;   // tokens (B*T)

// ---------------- GEMM tiling ----------------
constexpr int TM   = 32;            // token tile
constexpr int TN   = 128;           // output-feature tile
constexpr int KT   = 32;            // K step (one bf16 wmma)
constexpr int LDA  = 40;            // LDS stride (bf16): 80B rows, keeps 16B frag alignment
constexpr int LDB  = 40;
constexpr int MAXT = kN / TM + kE;  // worst-case per-expert-padded tile count = 136

typedef __attribute__((ext_vector_type(16))) __bf16 v16bf;
typedef __attribute__((ext_vector_type(8)))  __bf16 v8bf;
typedef __attribute__((ext_vector_type(8)))  float  v8f;

// ---------------- CDNA5 async-copy helpers ----------------
// GLOBAL_LOAD_ASYNC_TO_LDS_*: DMA global -> LDS, tracked by ASYNCcnt (ISA 10.x / 15.18.3).
// vdst = per-lane LDS byte address, vaddr = 64-bit global address (GV mode).
__device__ __forceinline__ uint32_t lds_off(const void* p) {
  return (uint32_t)(uintptr_t)p;  // generic LDS addr[31:0] == LDS offset (aperture rule)
}
__device__ __forceinline__ void async_b64(const __bf16* g, uint32_t l) {
  asm volatile("global_load_async_to_lds_b64 %0, %1, off"
               :: "v"(l), "v"((unsigned long long)(uintptr_t)g) : "memory");
}
__device__ __forceinline__ void async_b128(const __bf16* g, uint32_t l) {
  asm volatile("global_load_async_to_lds_b128 %0, %1, off"
               :: "v"(l), "v"((unsigned long long)(uintptr_t)g) : "memory");
}
template <int N>
__device__ __forceinline__ void wait_async() {
  asm volatile("s_wait_asynccnt %0" :: "n"(N) : "memory");
}

__device__ __forceinline__ v16bf ld_frag(const __bf16* p0, const __bf16* p1) {
  v8bf lo = *(const v8bf*)p0;
  v8bf hi = *(const v8bf*)p1;
  return __builtin_shufflevector(lo, hi, 0,1,2,3,4,5,6,7,8,9,10,11,12,13,14,15);
}

// ---------------- small helper kernels ----------------
__global__ void k_init(int* counts, int* cursors, unsigned int* zbuf) {
  int t = threadIdx.x;
  if (t < kE) { counts[t] = 0; cursors[t] = 0; }
  if (t < 32) zbuf[t] = 0u;   // 128B zero line for padded-row async sources
}

__global__ void k_cvt(const float* __restrict__ s, __bf16* __restrict__ d) {
  int i = (blockIdx.x * 256 + threadIdx.x) * 4;
  float4 v = *(const float4*)(s + i);
  d[i + 0] = (__bf16)v.x; d[i + 1] = (__bf16)v.y;
  d[i + 2] = (__bf16)v.z; d[i + 3] = (__bf16)v.w;
}

__global__ void k_gate(const float* __restrict__ x, const float* __restrict__ gw,
                       int* __restrict__ eid, int* __restrict__ counts) {
  int n = blockIdx.x * blockDim.x + threadIdx.x;
  if (n >= kN) return;
  const float* xr = x + (size_t)n * kD;
  float acc[kE];
#pragma unroll
  for (int e = 0; e < kE; ++e) acc[e] = 0.0f;
  for (int d = 0; d < kD; ++d) {
    float xv = xr[d];
#pragma unroll
    for (int e = 0; e < kE; ++e) acc[e] += xv * gw[e * kD + d];
  }
  int best = 0; float bv = acc[0];
#pragma unroll
  for (int e = 1; e < kE; ++e) if (acc[e] > bv) { bv = acc[e]; best = e; }
  eid[n] = best;                 // K=1 => softmax(top1) == 1.0, pure one-hot route
  atomicAdd(&counts[best], 1);
}

__global__ void k_scan(const int* __restrict__ counts, int* __restrict__ offs,
                       int* __restrict__ meta) {
  if (threadIdx.x != 0 || blockIdx.x != 0) return;
  int off = 0;
  for (int e = 0; e < kE; ++e) { offs[e] = off; off += counts[e]; }
  offs[kE] = off;
  int t = 0;
  for (int e = 0; e < kE; ++e) {
    int c = counts[e];
    for (int j = 0; j < c; j += TM) {
      meta[1 + t]            = e;
      meta[1 + MAXT + t]     = offs[e] + j;
      meta[1 + 2 * MAXT + t] = (c - j < TM) ? (c - j) : TM;
      ++t;
    }
  }
  meta[0] = t;
}

__global__ void k_scatter(const int* __restrict__ eid, const int* __restrict__ offs,
                          int* __restrict__ cursors, int* __restrict__ perm) {
  int n = blockIdx.x * blockDim.x + threadIdx.x;
  if (n >= kN) return;
  int e = eid[n];
  int p = atomicAdd(&cursors[e], 1);
  perm[offs[e] + p] = n;
}

// ---------------- stage 1 GEMM: h = act(x*W^T) [* (x*W3^T)] ----------------
// Double-buffered LDS, tiles filled by GLOBAL_LOAD_ASYNC_TO_LDS, consumed by WMMA.
template <bool EXPERT>
__global__ __launch_bounds__(256) void k_ffn_in(
    const __bf16* __restrict__ xb, const __bf16* __restrict__ Wg,
    const __bf16* __restrict__ Wu, const int* __restrict__ meta,
    const int* __restrict__ perm, const __bf16* __restrict__ zbuf,
    __bf16* __restrict__ hbuf) {
  int tile = blockIdx.x;
  int e = 0, qs, valid;
  if constexpr (EXPERT) {
    if (tile >= meta[0]) return;
    e     = meta[1 + tile];
    qs    = meta[1 + MAXT + tile];
    valid = meta[1 + 2 * MAXT + tile];
  } else {
    qs = tile * TM; valid = TM;
  }
  const int nb = blockIdx.y * TN;
  const __bf16* w1 = Wg + (EXPERT ? (size_t)e * kI * kD : 0);
  const __bf16* w3 = EXPERT ? (Wu + (size_t)e * kI * kD) : nullptr;

  __shared__ __bf16 sA[2][TM * LDA];
  __shared__ __bf16 sB1[2][TN * LDB];
  __shared__ __bf16 sB2[EXPERT ? 2 : 1][EXPERT ? TN * LDB : 8];

  const int tid  = threadIdx.x;
  const int lane = tid & 31;
  const int wv   = tid >> 5;
  const int wm   = wv >> 2;    // 0..1 (M wave)
  const int wn   = wv & 3;     // 0..3 (N wave)

  // A: one b64 (4 bf16) per thread; B: two b128 (2x8 bf16) per thread per matrix.
  const int ar = tid >> 3, ak = (tid & 7) * 4;
  const int br = tid >> 1, bk = (tid & 1) * 16;

  const bool a_ok = EXPERT ? (ar < valid) : true;
  const __bf16* abase_g;
  if constexpr (EXPERT) {
    abase_g = a_ok ? (xb + (size_t)perm[qs + ar] * kD + ak) : (zbuf + ak);
  } else {
    abase_g = xb + (size_t)(qs + ar) * kD + ak;
  }
  const uint32_t lA[2]  = { lds_off(&sA[0][ar * LDA + ak]),  lds_off(&sA[1][ar * LDA + ak]) };
  const uint32_t lB1[2] = { lds_off(&sB1[0][br * LDB + bk]), lds_off(&sB1[1][br * LDB + bk]) };
  const uint32_t lB2[2] = { lds_off(&sB2[0][EXPERT ? br * LDB + bk : 0]),
                            lds_off(&sB2[EXPERT ? 1 : 0][EXPERT ? br * LDB + bk : 0]) };

  auto load_tile = [&](int kt, int b) {
    async_b64(a_ok ? abase_g + kt : abase_g, lA[b]);
    const __bf16* g1 = w1 + (size_t)(nb + br) * kD + kt + bk;
    async_b128(g1,     lB1[b]);
    async_b128(g1 + 8, lB1[b] + 16);
    if constexpr (EXPERT) {
      const __bf16* g3 = w3 + (size_t)(nb + br) * kD + kt + bk;
      async_b128(g3,     lB2[b]);
      async_b128(g3 + 8, lB2[b] + 16);
    }
  };
  constexpr int NPF = EXPERT ? 5 : 3;   // async instrs per wave per tile

  v8f zz = {};
  v8f acc1[2] = {zz, zz};
  v8f acc3[2] = {zz, zz};

  const int m     = lane & 15;
  const int abase = (lane >> 4) * 16;   // A frag: k = abase + {0..15}
  const int kbb   = (lane >> 4) * 16;   // B frag: k = kbb + {0..15}

  constexpr int KITERS = kD / KT;
  load_tile(0, 0);
  for (int it = 0; it < KITERS; ++it) {
    const int cur = it & 1;
    __syncthreads();                       // everyone done reading buf[cur^1]
    if (it + 1 < KITERS) {
      load_tile((it + 1) * KT, cur ^ 1);   // prefetch next K tile
      wait_async<NPF>();                   // tile `it` landed; prefetch still in flight
    } else {
      wait_async<0>();
    }
    __syncthreads();                       // tile `it` visible to all waves
    const __bf16* pa = &sA[cur][(wm * 16 + m) * LDA + abase];
    v16bf af = ld_frag(pa, pa + 8);
#pragma unroll
    for (int c = 0; c < 2; ++c) {
      const int nrow = wn * 32 + c * 16 + m;
      const __bf16* pb = &sB1[cur][nrow * LDB + kbb];
      v16bf bf1 = ld_frag(pb, pb + 8);
      acc1[c] = __builtin_amdgcn_wmma_f32_16x16x32_bf16(
          false, af, false, bf1, (short)0, acc1[c], false, false);
      if constexpr (EXPERT) {
        const __bf16* pb2 = &sB2[cur][nrow * LDB + kbb];
        v16bf bf3 = ld_frag(pb2, pb2 + 8);
        acc3[c] = __builtin_amdgcn_wmma_f32_16x16x32_bf16(
            false, af, false, bf3, (short)0, acc3[c], false, false);
      }
    }
  }

  const int mofs = (lane >> 4) * 8;  // C/D layout: m = r + (lane>=16 ? 8 : 0)
  const int coll = lane & 15;
#pragma unroll
  for (int c = 0; c < 2; ++c) {
    const int col = nb + wn * 32 + c * 16 + coll;
#pragma unroll
    for (int r = 0; r < 8; ++r) {
      const int lr = wm * 16 + r + mofs;
      float v = acc1[c][r];
      float h;
      if constexpr (EXPERT) {
        h = (v / (1.0f + __expf(-v))) * acc3[c][r];          // SwiGLU
      } else {
        h = 0.5f * v * (1.0f + erff(v * 0.70710678118f));    // exact GELU
      }
      if (lr < valid) hbuf[(size_t)(qs + lr) * kI + col] = (__bf16)h;
    }
  }
}

// ---------------- stage 2 GEMM: y = h * W2^T ----------------
template <bool EXPERT>
__global__ __launch_bounds__(256) void k_ffn_out(
    const __bf16* __restrict__ hbuf, const __bf16* __restrict__ W2,
    const int* __restrict__ meta, const int* __restrict__ perm,
    const __bf16* __restrict__ zbuf, float* __restrict__ out) {
  int tile = blockIdx.x;
  int e = 0, qs, valid;
  if constexpr (EXPERT) {
    if (tile >= meta[0]) return;
    e     = meta[1 + tile];
    qs    = meta[1 + MAXT + tile];
    valid = meta[1 + 2 * MAXT + tile];
  } else {
    qs = tile * TM; valid = TM;
  }
  const int nb = blockIdx.y * TN;
  const __bf16* w2 = W2 + (EXPERT ? (size_t)e * kD * kI : 0);

  __shared__ __bf16 sA[2][TM * LDA];
  __shared__ __bf16 sB[2][TN * LDB];

  const int tid  = threadIdx.x;
  const int lane = tid & 31;
  const int wv   = tid >> 5;
  const int wm   = wv >> 2;
  const int wn   = wv & 3;

  const int ar = tid >> 3, ak = (tid & 7) * 4;
  const int br = tid >> 1, bk = (tid & 1) * 16;

  const bool a_ok = (ar < valid);
  const __bf16* abase_g = a_ok ? (hbuf + (size_t)(qs + ar) * kI + ak) : (zbuf + ak);
  const uint32_t lA[2] = { lds_off(&sA[0][ar * LDA + ak]), lds_off(&sA[1][ar * LDA + ak]) };
  const uint32_t lB[2] = { lds_off(&sB[0][br * LDB + bk]), lds_off(&sB[1][br * LDB + bk]) };

  auto load_tile = [&](int kt, int b) {
    async_b64(a_ok ? abase_g + kt : abase_g, lA[b]);
    const __bf16* g = w2 + (size_t)(nb + br) * kI + kt + bk;
    async_b128(g,     lB[b]);
    async_b128(g + 8, lB[b] + 16);
  };
  constexpr int NPF = 3;

  v8f zz = {};
  v8f acc[2] = {zz, zz};

  const int m     = lane & 15;
  const int abase = (lane >> 4) * 16;
  const int kbb   = (lane >> 4) * 16;

  constexpr int KITERS = kI / KT;
  load_tile(0, 0);
  for (int it = 0; it < KITERS; ++it) {
    const int cur = it & 1;
    __syncthreads();
    if (it + 1 < KITERS) {
      load_tile((it + 1) * KT, cur ^ 1);
      wait_async<NPF>();
    } else {
      wait_async<0>();
    }
    __syncthreads();
    const __bf16* pa = &sA[cur][(wm * 16 + m) * LDA + abase];
    v16bf af = ld_frag(pa, pa + 8);
#pragma unroll
    for (int c = 0; c < 2; ++c) {
      const int nrow = wn * 32 + c * 16 + m;
      const __bf16* pb = &sB[cur][nrow * LDB + kbb];
      v16bf bf = ld_frag(pb, pb + 8);
      acc[c] = __builtin_amdgcn_wmma_f32_16x16x32_bf16(
          false, af, false, bf, (short)0, acc[c], false, false);
    }
  }

  const int mofs = (lane >> 4) * 8;
  const int coll = lane & 15;
#pragma unroll
  for (int c = 0; c < 2; ++c) {
    const int col = nb + wn * 32 + c * 16 + coll;
#pragma unroll
    for (int r = 0; r < 8; ++r) {
      const int lr = wm * 16 + r + mofs;
      if (lr >= valid) continue;
      float v = acc[c][r];
      if constexpr (EXPERT) {
        int tok = perm[qs + lr];
        out[(size_t)tok * kD + col] = v;        // each token routed exactly once
      } else {
        out[(size_t)(qs + lr) * kD + col] += v; // shared-expert accumulate
      }
    }
  }
}

// ---------------- launcher ----------------
extern "C" void kernel_launch(void* const* d_in, const int* in_sizes, int n_in,
                              void* d_out, int out_size, void* d_ws, size_t ws_size,
                              hipStream_t stream) {
  (void)in_sizes; (void)n_in; (void)out_size; (void)ws_size;
  const float* x   = (const float*)d_in[0];
  const float* gw  = (const float*)d_in[1];
  const float* w1  = (const float*)d_in[2];
  const float* w2  = (const float*)d_in[3];
  const float* w3  = (const float*)d_in[4];
  const float* sw1 = (const float*)d_in[5];
  const float* sw2 = (const float*)d_in[6];
  float* out = (float*)d_out;

  char* p = (char*)d_ws;
  auto take = [&](size_t b) { char* r = p; p += (b + 255) & ~(size_t)255; return r; };
  int* eid           = (int*)take((size_t)kN * 4);
  int* counts        = (int*)take(kE * 4);
  int* cursors       = (int*)take(kE * 4);
  int* offs          = (int*)take((kE + 1) * 4);
  int* meta          = (int*)take((1 + 3 * MAXT) * 4);
  int* perm          = (int*)take((size_t)kN * 4);
  unsigned int* zb   = (unsigned int*)take(128);
  __bf16* zbuf       = (__bf16*)zb;
  __bf16* xb         = (__bf16*)take((size_t)kN * kD * 2);
  __bf16* hbuf       = (__bf16*)take((size_t)kN * kI * 2);   // expert h, then shared h
  __bf16* wb1        = (__bf16*)take((size_t)kE * kI * kD * 2);
  __bf16* wb3        = (__bf16*)take((size_t)kE * kI * kD * 2);
  __bf16* wb2        = (__bf16*)take((size_t)kE * kD * kI * 2);
  __bf16* sb1        = (__bf16*)take((size_t)kI * kD * 2);
  __bf16* sb2        = (__bf16*)take((size_t)kD * kI * 2);

  k_init<<<1, 32, 0, stream>>>(counts, cursors, zb);
  // fp32 -> bf16 once per launch (~165MB traffic ≈ 7us at 23.3 TB/s; compute-bound overall)
  k_cvt<<<(kN * kD) / 1024, 256, 0, stream>>>(x, xb);
  k_cvt<<<(kE * kI * kD) / 1024, 256, 0, stream>>>(w1, wb1);
  k_cvt<<<(kE * kI * kD) / 1024, 256, 0, stream>>>(w3, wb3);
  k_cvt<<<(kE * kD * kI) / 1024, 256, 0, stream>>>(w2, wb2);
  k_cvt<<<(kI * kD) / 1024, 256, 0, stream>>>(sw1, sb1);
  k_cvt<<<(kD * kI) / 1024, 256, 0, stream>>>(sw2, sb2);

  k_gate<<<kN / 256, 256, 0, stream>>>(x, gw, eid, counts);
  k_scan<<<1, 1, 0, stream>>>(counts, offs, meta);
  k_scatter<<<kN / 256, 256, 0, stream>>>(eid, offs, cursors, perm);

  // routed experts (grouped by argmax expert; top-1 => combine weight 1.0)
  k_ffn_in<true ><<<dim3(MAXT, kI / TN), 256, 0, stream>>>(xb, wb1, wb3, meta, perm, zbuf, hbuf);
  k_ffn_out<true ><<<dim3(MAXT, kD / TN), 256, 0, stream>>>(hbuf, wb2, meta, perm, zbuf, out);
  // shared expert (all tokens), accumulates into out; reuses hbuf (stream-ordered)
  k_ffn_in<false><<<dim3(kN / TM, kI / TN), 256, 0, stream>>>(xb, sb1, nullptr, nullptr, nullptr, zbuf, hbuf);
  k_ffn_out<false><<<dim3(kN / TM, kD / TN), 256, 0, stream>>>(hbuf, sb2, nullptr, nullptr, zbuf, out);
}
